// Pixel2MeshModel_37538014167434
// MI455X (gfx1250) — matrix operand order repaired and checked by reading, not compiled, not run on hardware
//
#include <hip/hip_runtime.h>

typedef __bf16 bf16_t;
typedef __attribute__((ext_vector_type(16))) __bf16 v16bf;
typedef __attribute__((ext_vector_type(8)))  __bf16 v8bf;
typedef __attribute__((ext_vector_type(8)))  float  v8f;

#define NVERT 10000
#define NPAD  10016      // K dim of adj GEMM, padded to multiple of 32
#define BATCH 4
#define HDIM  128
#define VT    625        // NVERT / 16 vertex tiles
#define FIN   963
#define FINP  992        // FIN padded to multiple of 32

union ABfrag { v16bf v; v8bf h[2]; };

// ---------------------------------------------------------------------------
// Conversion kernels
// ---------------------------------------------------------------------------

// fp32 [rows][K] -> bf16 [rows][Kpad] (zero pad)
__global__ void k_cvt_pad(const float* __restrict__ src, bf16_t* __restrict__ dst,
                          int K, int Kpad) {
  int c = blockIdx.x * blockDim.x + threadIdx.x;
  int r = blockIdx.y;
  if (c >= Kpad) return;
  float v = (c < K) ? src[(size_t)r * K + c] : 0.f;
  dst[(size_t)r * Kpad + c] = (bf16_t)v;
}

// fp32 W [K][H] -> bf16 Wt [Hpad][Kpad] (transposed, zero pad)
__global__ void k_cvt_wT(const float* __restrict__ src, bf16_t* __restrict__ dst,
                         int K, int H, int Kpad) {
  int k = blockIdx.x * blockDim.x + threadIdx.x;
  int h = blockIdx.y;
  if (k >= Kpad) return;
  float v = (k < K && h < H) ? src[(size_t)k * H + h] : 0.f;
  dst[(size_t)h * Kpad + k] = (bf16_t)v;
}

// zero K-pad columns [NVERT, NPAD) of xwT for all 4*128 rows (written once;
// transform kernels only ever write columns < NVERT)
__global__ void k_zero_xwt_pad(bf16_t* __restrict__ xwT) {
  int t = blockIdx.x * blockDim.x + threadIdx.x;
  if (t >= BATCH * HDIM * (NPAD - NVERT)) return;
  int row = t / (NPAD - NVERT);
  int c   = t % (NPAD - NVERT);
  xwT[(size_t)row * NPAD + NVERT + c] = (bf16_t)0.f;
}

// ---------------------------------------------------------------------------
// Feature transform: per wave, one (batch, vertex-tile, h-tile).
//   acc0 = X @ W   -> written transposed as bf16 into xwT  [b][h][NPAD]
//   acc1 = X @ LW  -> + bias, written fp32 into skip       [b*N + v][Hs]
// ---------------------------------------------------------------------------
__global__ void __launch_bounds__(256)
k_xform_wmma(const bf16_t* __restrict__ Xb,   // [B*NVERT][Kpad] bf16
             const bf16_t* __restrict__ Wt,   // [Hpad][Kpad]
             const bf16_t* __restrict__ LWt,  // [Hpad][Kpad]
             const float*  __restrict__ bias, // [Hreal]
             float*        __restrict__ skip, // [B*NVERT][Hs]
             bf16_t*       __restrict__ xwT,  // [B][HDIM][NPAD]
             int Kpad, int htiles, int Hs, int Hreal) {
  int gid  = blockIdx.x * blockDim.x + threadIdx.x;
  int wv   = gid >> 5, lane = gid & 31;
  int total = BATCH * VT * htiles;
  if (wv >= total) return;                 // whole-wave exit: EXEC stays all-1s
  int b  = wv / (VT * htiles);
  int r  = wv % (VT * htiles);
  int vt = r / htiles, ht = r % htiles;
  int l  = lane & 15, kh = lane >> 4;

  const bf16_t* arow = Xb  + (size_t)(b * NVERT + vt * 16 + l) * Kpad;
  const bf16_t* brow = Wt  + (size_t)(ht * 16 + l) * Kpad;
  const bf16_t* crow = LWt + (size_t)(ht * 16 + l) * Kpad;

  v8f zero = {0.f, 0.f, 0.f, 0.f, 0.f, 0.f, 0.f, 0.f};
  v8f acc0 = zero, acc1 = zero;

  for (int k0 = 0; k0 < Kpad; k0 += 32) {
    ABfrag a, bw, bl;
    a.h[0]  = *(const v8bf*)(arow + k0 + kh * 16);       // A: K 0-7 / 16-23
    a.h[1]  = *(const v8bf*)(arow + k0 + kh * 16 + 8);   //    K 8-15 / 24-31
    bw.h[0] = *(const v8bf*)(brow + k0 + kh * 16);       // B: K 0-15 / 16-31
    bw.h[1] = *(const v8bf*)(brow + k0 + kh * 16 + 8);
    bl.h[0] = *(const v8bf*)(crow + k0 + kh * 16);
    bl.h[1] = *(const v8bf*)(crow + k0 + kh * 16 + 8);
    acc0 = __builtin_amdgcn_wmma_f32_16x16x32_bf16(false, a.v, false, bw.v,
                                                   (short)0, acc0, false, false);
    acc1 = __builtin_amdgcn_wmma_f32_16x16x32_bf16(false, a.v, false, bl.v,
                                                   (short)0, acc1, false, false);
  }

  int h  = ht * 16 + l;                       // lane-owned output column
  float bv = (h < Hreal) ? bias[h] : 0.f;
  int vbase = vt * 16 + kh * 8;               // 8 consecutive vertex rows
  v8bf xw;
#pragma unroll
  for (int g = 0; g < 8; ++g) {
    skip[(size_t)(b * NVERT + vbase + g) * Hs + h] = acc1[g] + bv;
    xw[g] = (bf16_t)acc0[g];
  }
  *(v8bf*)(xwT + (size_t)(b * HDIM + h) * NPAD + vbase) = xw;   // one b128 store
}

// ---------------------------------------------------------------------------
// Adjacency GEMM: per wave one (batch, vertex-tile) x NT h-tiles.
//   Y = adj @ xW + skip ; mode 0: relu -> outF,outB
//                         mode 1: (xold + relu)*0.5 -> outF,outB (residual)
//                         mode 2: plain -> outF (masked cols, output layer)
// ---------------------------------------------------------------------------
template <int NT>
__global__ void __launch_bounds__(256)
k_adj_wmma(const bf16_t* __restrict__ adjB,  // [NVERT][NPAD]
           const bf16_t* __restrict__ xwT,   // [B][HDIM][NPAD]
           const float*  __restrict__ skip,  // [B*NVERT][Hs]
           const float*  __restrict__ xold,  // [B*NVERT][HDIM] or null
           float*        __restrict__ outF,  // stride sF
           bf16_t*       __restrict__ outB,  // [B*NVERT][HDIM] or null
           int Hs, int sF, int ncols, int mode) {
  int gid = blockIdx.x * blockDim.x + threadIdx.x;
  int wv  = gid >> 5, lane = gid & 31;
  if (wv >= BATCH * VT) return;              // whole-wave exit
  int b = wv / VT, mt = wv % VT;
  int l = lane & 15, kh = lane >> 4;

  const bf16_t* arow  = adjB + (size_t)(mt * 16 + l) * NPAD;
  const bf16_t* bbase = xwT  + (size_t)(b * HDIM + l) * NPAD;

  v8f zero = {0.f, 0.f, 0.f, 0.f, 0.f, 0.f, 0.f, 0.f};
  v8f acc[NT];
#pragma unroll
  for (int i = 0; i < NT; ++i) acc[i] = zero;

  for (int k0 = 0; k0 < NPAD; k0 += 32) {
    __builtin_prefetch(arow + k0 + 512, 0, 1);   // global_prefetch_b8
    ABfrag a;
    a.h[0] = *(const v8bf*)(arow + k0 + kh * 16);
    a.h[1] = *(const v8bf*)(arow + k0 + kh * 16 + 8);
#pragma unroll
    for (int nt = 0; nt < NT; ++nt) {
      const bf16_t* br = bbase + (size_t)nt * 16 * NPAD;
      ABfrag bw;
      bw.h[0] = *(const v8bf*)(br + k0 + kh * 16);
      bw.h[1] = *(const v8bf*)(br + k0 + kh * 16 + 8);
      acc[nt] = __builtin_amdgcn_wmma_f32_16x16x32_bf16(false, a.v, false, bw.v,
                                                        (short)0, acc[nt], false, false);
    }
  }

  int vbase = mt * 16 + kh * 8;
#pragma unroll
  for (int nt = 0; nt < NT; ++nt) {
    int h = nt * 16 + l;
#pragma unroll
    for (int g = 0; g < 8; ++g) {
      size_t gi = (size_t)(b * NVERT) + vbase + g;
      float y = acc[nt][g] + skip[gi * Hs + h];
      if (mode == 0)      y = fmaxf(y, 0.f);
      else if (mode == 1) y = 0.5f * (xold[gi * HDIM + h] + fmaxf(y, 0.f));
      if (h < ncols) outF[gi * sF + h] = y;
      if (outB)      outB[gi * HDIM + h] = (bf16_t)y;
    }
  }
}

// ---------------------------------------------------------------------------
// Workspace layout (all chunks 256B aligned by construction)
// ---------------------------------------------------------------------------
static constexpr size_t SZ_ADJB = (size_t)NVERT * NPAD * 2;          // 200,320,000
static constexpr size_t SZ_XB   = (size_t)BATCH * NVERT * FINP * 2;  //  79,360,000
static constexpr size_t SZ_XWT  = (size_t)BATCH * HDIM * NPAD * 2;   //  10,256,384
static constexpr size_t SZ_ACT  = (size_t)BATCH * NVERT * HDIM * 4;  //  20,480,000
static constexpr size_t O_ADJB = 0;
static constexpr size_t O_XB   = O_ADJB + SZ_ADJB;
static constexpr size_t O_XWT  = O_XB + SZ_XB;
static constexpr size_t O_SKIP = O_XWT + SZ_XWT;
static constexpr size_t O_X    = O_SKIP + SZ_ACT;
static constexpr size_t O_H1   = O_X + SZ_ACT;
static constexpr size_t O_WTS  = O_H1 + SZ_ACT;
static constexpr size_t SZ_WTS = 2 * 262144 + 12 * 32768 + 2 * 4096; // wide + res + out
static constexpr size_t WS_NEED = O_WTS + SZ_WTS;

extern "C" void kernel_launch(void* const* d_in, const int* in_sizes, int n_in,
                              void* d_out, int out_size, void* d_ws, size_t ws_size,
                              hipStream_t stream) {
  (void)in_sizes; (void)n_in; (void)out_size;
  if (ws_size < WS_NEED) return;

  const float* shape = (const float*)d_in[0];
  const float* adj   = (const float*)d_in[1];
  const float* in_w  = (const float*)d_in[2];
  const float* in_lw = (const float*)d_in[3];
  const float* in_b  = (const float*)d_in[4];
  const float* rw1   = (const float*)d_in[5];
  const float* rlw1  = (const float*)d_in[6];
  const float* rb1   = (const float*)d_in[7];
  const float* rw2   = (const float*)d_in[8];
  const float* rlw2  = (const float*)d_in[9];
  const float* rb2   = (const float*)d_in[10];
  const float* ow    = (const float*)d_in[11];
  const float* olw   = (const float*)d_in[12];
  const float* ob    = (const float*)d_in[13];

  char* ws = (char*)d_ws;
  bf16_t* adjB = (bf16_t*)(ws + O_ADJB);
  bf16_t* xb   = (bf16_t*)(ws + O_XB);     // bf16 activations (stride FINP then HDIM)
  bf16_t* xwT  = (bf16_t*)(ws + O_XWT);
  float*  skip = (float*)(ws + O_SKIP);
  float*  x    = (float*)(ws + O_X);
  float*  h1   = (float*)(ws + O_H1);
  bf16_t* w_in   = (bf16_t*)(ws + O_WTS);
  bf16_t* w_inl  = (bf16_t*)(ws + O_WTS + 262144);
  bf16_t* w_res  = (bf16_t*)(ws + O_WTS + 2 * 262144);    // 12 slots of 16384 elems
  bf16_t* w_out  = (bf16_t*)(ws + O_WTS + 2 * 262144 + 12 * 32768);
  bf16_t* w_outl = w_out + 2048;

  float* outY = (float*)d_out;                       // x_out: [4][10000][3]
  float* outX = outY + (size_t)BATCH * NVERT * 3;    // x:     [4][10000][128]

  dim3 t256(256);

  // --- conversions -------------------------------------------------------
  k_cvt_pad<<<dim3((NPAD + 255) / 256, NVERT), t256, 0, stream>>>(adj, adjB, NVERT, NPAD);
  k_cvt_pad<<<dim3((FINP + 255) / 256, BATCH * NVERT), t256, 0, stream>>>(shape, xb, FIN, FINP);
  k_zero_xwt_pad<<<(BATCH * HDIM * (NPAD - NVERT) + 255) / 256, t256, 0, stream>>>(xwT);

  k_cvt_wT<<<dim3(FINP / 256 + 1, HDIM), t256, 0, stream>>>(in_w,  w_in,  FIN, HDIM, FINP);
  k_cvt_wT<<<dim3(FINP / 256 + 1, HDIM), t256, 0, stream>>>(in_lw, w_inl, FIN, HDIM, FINP);
  for (int i = 0; i < 3; ++i) {
    const float* srcs[4] = {rw1 + (size_t)i * HDIM * HDIM, rlw1 + (size_t)i * HDIM * HDIM,
                            rw2 + (size_t)i * HDIM * HDIM, rlw2 + (size_t)i * HDIM * HDIM};
    for (int j = 0; j < 4; ++j)
      k_cvt_wT<<<dim3(1, HDIM), t256, 0, stream>>>(srcs[j], w_res + (size_t)(i * 4 + j) * 16384,
                                                   HDIM, HDIM, HDIM);
  }
  k_cvt_wT<<<dim3(1, 16), t256, 0, stream>>>(ow,  w_out,  HDIM, 3, HDIM);
  k_cvt_wT<<<dim3(1, 16), t256, 0, stream>>>(olw, w_outl, HDIM, 3, HDIM);

  const int blkAdj = (BATCH * VT * 32 + 255) / 256;            // 313
  const int blkXf8 = (BATCH * VT * 8 * 32 + 255) / 256;        // 2500
  const int blkXf1 = (BATCH * VT * 1 * 32 + 255) / 256;        // 313

  // --- input layer: x = relu(gcn(shape_verts)) --------------------------
  k_xform_wmma<<<blkXf8, t256, 0, stream>>>(xb, w_in, w_inl, in_b, skip, xwT,
                                            FINP, 8, HDIM, HDIM);
  k_adj_wmma<8><<<blkAdj, t256, 0, stream>>>(adjB, xwT, skip, nullptr, x, xb,
                                             HDIM, HDIM, HDIM, 0);

  // --- residual blocks ---------------------------------------------------
  for (int i = 0; i < 3; ++i) {
    bf16_t* s = w_res + (size_t)(i * 4) * 16384;
    // h1 = relu(gcn(x))
    k_xform_wmma<<<blkXf8, t256, 0, stream>>>(xb, s, s + 16384, rb1 + (size_t)i * HDIM,
                                              skip, xwT, HDIM, 8, HDIM, HDIM);
    k_adj_wmma<8><<<blkAdj, t256, 0, stream>>>(adjB, xwT, skip, nullptr, h1, xb,
                                               HDIM, HDIM, HDIM, 0);
    // x = (x + relu(gcn(h1))) * 0.5   (last block writes fp32 x straight to d_out)
    float* xdst = (i == 2) ? outX : x;
    k_xform_wmma<<<blkXf8, t256, 0, stream>>>(xb, s + 2 * 16384, s + 3 * 16384,
                                              rb2 + (size_t)i * HDIM, skip, xwT,
                                              HDIM, 8, HDIM, HDIM);
    k_adj_wmma<8><<<blkAdj, t256, 0, stream>>>(adjB, xwT, skip, x, xdst, xb,
                                               HDIM, HDIM, HDIM, 1);
  }

  // --- output layer: x_out = gcn(x), H_out = 3 (padded to one 16-tile) ---
  k_xform_wmma<<<blkXf1, t256, 0, stream>>>(xb, w_out, w_outl, ob, skip, xwT,
                                            HDIM, 1, 16, 3);
  k_adj_wmma<1><<<blkAdj, t256, 0, stream>>>(adjB, xwT, skip, nullptr, outY, nullptr,
                                             16, 3, 3, 2);
}
